// KCNet_42571715837960
// MI455X (gfx1250) — matrix operand order repaired and checked by reading, not compiled, not run on hardware
//
#include <hip/hip_runtime.h>
#include <hip/hip_bf16.h>

#define NPTS 262144
#define KNN  16
#define AS   68     // f32 act tile stride: 64 + 4 pad (bank-conflict free row-parallel reads)
#define HCS2 200    // f16 tile stride in halves: 192 + 8 pad (36m mod 64 permutation, B128-safe)

typedef __attribute__((ext_vector_type(2)))  float    v2f;
typedef __attribute__((ext_vector_type(8)))  float    v8f;
typedef __attribute__((ext_vector_type(16))) _Float16 v16h;
typedef __attribute__((ext_vector_type(8)))  _Float16 h8;
typedef __attribute__((ext_vector_type(2)))  _Float16 h2;
typedef __attribute__((ext_vector_type(4)))  unsigned uint32x4;
typedef __attribute__((ext_vector_type(4)))  int      int32x4;
typedef __attribute__((ext_vector_type(8)))  int      int32x8;

#if defined(__has_builtin)
#if __has_builtin(__builtin_amdgcn_tensor_load_to_lds) && \
    __has_builtin(__builtin_amdgcn_s_wait_tensorcnt)
#define HAVE_TDM 1
#endif
#endif

__device__ __forceinline__ v8f wmma_k4(v2f a, v2f b, v8f c) {
  // exact fp32 on the matrix pipe for the small per-point layers
  return __builtin_amdgcn_wmma_f32_16x16x4_f32(false, a, false, b, (short)0, c,
                                               false, false);
}

// ---- fp32 16xK @ W^T tile GEMM + relu; optional LDS f32 out, global f16 out
__device__ __forceinline__ void gemm16_relu(
    const float* __restrict__ aIn,     // LDS [16][AS]
    const float* __restrict__ W,       // global [outD][kD] row-major
    float*       __restrict__ aOut,    // LDS [16][AS] or nullptr
    _Float16*    __restrict__ gOut,    // global f16 row base or nullptr
    int gStride, int kD, int outTiles, int lane)
{
  const int m  = lane & 15;
  const int hi = lane >> 4;
  const int n  = lane & 15;
  for (int nt = 0; nt < outTiles; ++nt) {
    v8f c = {};
    const float* wrow = W + (size_t)(nt * 16 + n) * kD;
    for (int k0 = 0; k0 < kD; k0 += 4) {
      const int ka = k0 + 2 * hi;
      v2f a; a.x = aIn[m * AS + ka]; a.y = aIn[m * AS + ka + 1];
      v2f b; b.x = wrow[ka];         b.y = wrow[ka + 1];
      c = wmma_k4(a, b, c);
    }
#pragma unroll
    for (int v = 0; v < 8; ++v) {
      const float r   = c[v] > 0.f ? c[v] : 0.f;
      const int   row = v + 8 * hi;
      const int   col = nt * 16 + n;
      if (aOut) aOut[row * AS + col] = r;
      if (gOut) gOut[(size_t)row * gStride + col] = (_Float16)r;
    }
  }
}

// ---------------- Kernel A: fused per-point MLP (mlp1..mlp4), fp32 WMMA ----
__global__ __launch_bounds__(256) void kc_point_mlp(
    const float* __restrict__ x,   // [N,3]
    const float* __restrict__ w1,  // [64,3]
    const float* __restrict__ w2,  // [64,64]
    const float* __restrict__ w3,  // [64,64]
    const float* __restrict__ w4,  // [128,64]
    _Float16* __restrict__ h_out,  // [N,64]  f16 (only consumed by gather)
    _Float16* __restrict__ h4_out) // [N,128] f16 (only consumed by concat)
{
  extern __shared__ float smem[];
  const int wid  = threadIdx.x >> 5;
  const int lane = threadIdx.x & 31;
  float* actA = smem + wid * (2 * 16 * AS);
  float* actB = actA + 16 * AS;
  const int pt0 = (blockIdx.x * 8 + wid) * 16;

  // stage x tile [16][3] zero-padded to K=4
#pragma unroll
  for (int t = 0; t < 2; ++t) {
    const int e = lane + t * 32;
    const int m = e >> 2, c = e & 3;
    actA[m * AS + c] = (c < 3) ? x[(size_t)(pt0 + m) * 3 + c] : 0.f;
  }

  // layer1: relu(x @ w1^T), K padded 3->4 (branch-free B col3 = 0)
  {
    const int m = lane & 15, hi = lane >> 4, n = lane & 15;
    for (int nt = 0; nt < 4; ++nt) {
      v8f c = {};
      v2f a; a.x = actA[m * AS + 2 * hi]; a.y = actA[m * AS + 2 * hi + 1];
      const float* wrow = w1 + (size_t)(nt * 16 + n) * 3;
      const float wlo = wrow[2 * hi];   // k=0 (hi=0) or k=2 (hi=1)
      const float wmid = wrow[1];       // always valid
      v2f b; b.x = wlo; b.y = hi ? 0.f : wmid;
      c = wmma_k4(a, b, c);
#pragma unroll
      for (int v = 0; v < 8; ++v) {
        const float r = c[v] > 0.f ? c[v] : 0.f;
        actB[(v + 8 * hi) * AS + nt * 16 + n] = r;
      }
    }
  }
  gemm16_relu(actB, w2, actA, h_out + (size_t)pt0 * 64, 64, 64, 4, lane);
  gemm16_relu(actA, w3, actB, nullptr, 0, 64, 4, lane);
  gemm16_relu(actB, w4, nullptr, h4_out + (size_t)pt0 * 128, 128, 64, 8, lane);
}

// -------- Kernel C: gather-max + concat + f16 h5 GEMM + global max ---------
__global__ __launch_bounds__(256) void kc_graph_gemm_max(
    const _Float16* __restrict__ hH,      // [N,64]  f16
    const _Float16* __restrict__ h4H,     // [N,128] f16
    const int*      __restrict__ indices, // [N*16]
    const _Float16* __restrict__ w5h,     // [1024,192] f16
    float* __restrict__ gmax)             // [1024], uint-max protocol (>= 0)
{
  extern __shared__ char smemc[];
  _Float16* hcH   = (_Float16*)smemc;          // 8 * 16 * HCS2
  _Float16* w5s0  = hcH + 8 * 16 * HCS2;       // 16 * HCS2 (double buffer A)
  _Float16* w5s1  = w5s0 + 16 * HCS2;          // 16 * HCS2 (double buffer B)
  float*    gpart = (float*)(w5s1 + 16 * HCS2);// 1024 f32
  const int tid  = threadIdx.x;
  const int wid  = tid >> 5;
  const int lane = tid & 31;
  _Float16* myhc = hcH + wid * 16 * HCS2;
  const int pt0  = (blockIdx.x * 8 + wid) * 16;

#ifdef HAVE_TDM
  // Tensor Data Mover: DMA the 16x128-f16 h4 tile into hc cols 64..191
  // (LDS row stride 400B = 64 payload DWORDs + 36 pad DWORDs), overlapping
  // the neighbor gather below. D# per CDNA5 ISA ch.8.
  {
    const unsigned long long ga =
        (unsigned long long)(size_t)(h4H + (size_t)pt0 * 128);
    const unsigned ldsa = (unsigned)(size_t)(const void*)&myhc[64];
    uint32x4 g0;
    g0[0] = 1u;                                   // count=1 (valid descriptor)
    g0[1] = ldsa;                                 // lds_addr (bytes)
    g0[2] = (unsigned)(ga & 0xFFFFFFFFu);         // global_addr[31:0]
    g0[3] = ((unsigned)(ga >> 32) & 0x01FFFFFFu)  // global_addr[56:32]
            | 0x80000000u;                        // type=2 ("image")
    int32x8 g1;
    g1[0] = (int)((1u << 16)      // data_size=1 (2B)
                | (1u << 20)      // pad_enable
                | (5u << 22)      // pad_interval: 64 DWORDs (=256B row payload)
                | (35u << 25));   // pad_amount: 36 DWORDs (=144B -> 400B stride)
    g1[1] = (int)(128u << 16);    // tensor_dim0 = 128 elements
    g1[2] = (int)(16u << 16);     // tensor_dim1 = 16 rows
    g1[3] = (int)(128u << 16);    // tile_dim0 = 128
    g1[4] = 16;                   // tile_dim1 = 16
    g1[5] = 128;                  // tensor_dim0_stride = 128 elements
    g1[6] = 0;
    g1[7] = 0;
    const int32x4 z4 = {0, 0, 0, 0};
#if __clang_major__ >= 23
    const int32x8 z8 = {0, 0, 0, 0, 0, 0, 0, 0};
    __builtin_amdgcn_tensor_load_to_lds(g0, g1, z4, z4, z8, 0);
#else
    __builtin_amdgcn_tensor_load_to_lds(g0, g1, z4, z4, 0);
#endif
  }
#endif

  // neighbor gather + segment max -> cols 0..63 (rows 128B contiguous)
  for (int m = 0; m < 16; ++m) {
    const int* nb = indices + (size_t)(pt0 + m) * KNN;
    h2 acc = {};                                   // h >= 0 (post-ReLU)
    for (int k = 0; k < KNN; ++k) {
      const int id = nb[k];
      if (k + 1 < KNN) __builtin_prefetch(hH + (size_t)nb[k + 1] * 64, 0, 0);
      const h2 hv = *(const h2*)(hH + (size_t)id * 64 + lane * 2);
      acc.x = hv.x > acc.x ? hv.x : acc.x;
      acc.y = hv.y > acc.y ? hv.y : acc.y;
    }
    *(h2*)(&myhc[m * HCS2 + lane * 2]) = acc;
  }

#ifdef HAVE_TDM
  __builtin_amdgcn_s_wait_tensorcnt((short)0);
#else
  typedef __attribute__((ext_vector_type(4))) _Float16 h4v;
  for (int m = 0; m < 16; ++m) {
    const h4v hv = *(const h4v*)(h4H + (size_t)(pt0 + m) * 128 + lane * 4);
    *(h4v*)(&myhc[m * HCS2 + 64 + lane * 4]) = hv;
  }
#endif
  for (int i = tid; i < 1024; i += 256) gpart[i] = 0.f;

  // cooperative w5 tile staging (16 x 192 f16 = 6 KB), double buffered
  auto stage = [&](int nt, _Float16* dst) {
    const unsigned* src = (const unsigned*)(w5h + (size_t)nt * 16 * 192);
#pragma unroll
    for (int t = 0; t < 6; ++t) {
      const int i   = tid + t * 256;   // dword 0..1535
      const int e   = i * 2;
      const int row = e / 192, col = e % 192;
      *(unsigned*)(&dst[row * HCS2 + col]) = src[i];
    }
  };

  stage(0, w5s0);
  __syncthreads();

  const int m = lane & 15, hi = lane >> 4, n = lane & 15;
  const _Float16* arow = &myhc[m * HCS2];
  for (int nt = 0; nt < 64; ++nt) {
    if (nt + 1 < 64) stage(nt + 1, (nt & 1) ? w5s0 : w5s1);
    const _Float16* brow = ((nt & 1) ? w5s1 : w5s0) + n * HCS2;

    v8f c = {};
#pragma unroll
    for (int k0 = 0; k0 < 192; k0 += 32) {
      // A frag: two contiguous 16B runs; B frag: two contiguous 16B runs
      const h8 a0 = *(const h8*)(arow + k0 + 8 * hi);
      const h8 a1 = *(const h8*)(arow + k0 + 16 + 8 * hi);
      const h8 b0 = *(const h8*)(brow + k0 + 8 * hi);
      const h8 b1 = *(const h8*)(brow + k0 + 16 + 8 * hi);
      const v16h a = __builtin_shufflevector(a0, a1, 0, 1, 2, 3, 4, 5, 6, 7,
                                             8, 9, 10, 11, 12, 13, 14, 15);
      const v16h b = __builtin_shufflevector(b0, b1, 0, 1, 2, 3, 4, 5, 6, 7,
                                             8, 9, 10, 11, 12, 13, 14, 15);
      c = __builtin_amdgcn_wmma_f32_16x16x32_f16(false, a, false, b, (short)0,
                                                 c, false, false);
    }
    // relu + max over this wave's 16 points, fold into block partial
    float r = 0.f;                               // implicit ReLU
#pragma unroll
    for (int v = 0; v < 8; ++v) r = fmaxf(r, c[v]);
    r = fmaxf(r, __shfl_xor(r, 16, 32));
    if (lane < 16)
      atomicMax((unsigned int*)&gpart[nt * 16 + n], __float_as_uint(r));
    __syncthreads();   // stage(nt+1) visible; compute(nt) done before restage
  }

  for (int i = tid; i < 1024; i += 256)
    atomicMax((unsigned int*)&gmax[i], __float_as_uint(gpart[i]));
}

// ---------------- Kernel D: classifier head (1x1024 -> 40), fp32 ----------
__global__ __launch_bounds__(512) void kc_head(
    const float* __restrict__ gmax, const float* __restrict__ wf1,
    const float* __restrict__ wf2,  const float* __restrict__ wf3,
    float* __restrict__ out)
{
  __shared__ float g[1024], f1[512], f2[256];
  const int t = threadIdx.x;
  g[t] = gmax[t]; g[t + 512] = gmax[t + 512];
  __syncthreads();
  {
    float acc = 0.f;
    const float* w = wf1 + (size_t)t * 1024;
    for (int k = 0; k < 1024; ++k) acc += g[k] * w[k];
    f1[t] = acc > 0.f ? acc : 0.f;
  }
  __syncthreads();
  if (t < 256) {
    float acc = 0.f;
    const float* w = wf2 + (size_t)t * 512;
    for (int k = 0; k < 512; ++k) acc += f1[k] * w[k];
    f2[t] = acc > 0.f ? acc : 0.f;
  }
  __syncthreads();
  if (t < 40) {
    float acc = 0.f;
    const float* w = wf3 + (size_t)t * 256;
    for (int k = 0; k < 256; ++k) acc += f2[k] * w[k];
    out[t] = acc;
  }
}

__global__ void kc_zero(float* p, int n) {
  const int i = blockIdx.x * blockDim.x + threadIdx.x;
  if (i < n) p[i] = 0.f;
}

__global__ void kc_f32_to_f16(const float* __restrict__ src,
                              _Float16* __restrict__ dst, int n) {
  const int i = blockIdx.x * blockDim.x + threadIdx.x;
  if (i < n) dst[i] = (_Float16)src[i];
}

extern "C" void kernel_launch(void* const* d_in, const int* in_sizes, int n_in,
                              void* d_out, int out_size, void* d_ws, size_t ws_size,
                              hipStream_t stream) {
  const float* x       = (const float*)d_in[0];
  // d_in[1] = indptr: fixed degree K=16, unused
  const int*   indices = (const int*)d_in[2];
  const float* w1  = (const float*)d_in[3];
  const float* w2  = (const float*)d_in[4];
  const float* w3  = (const float*)d_in[5];
  const float* w4  = (const float*)d_in[6];
  const float* w5  = (const float*)d_in[7];
  const float* wf1 = (const float*)d_in[8];
  const float* wf2 = (const float*)d_in[9];
  const float* wf3 = (const float*)d_in[10];
  float* out = (float*)d_out;

  _Float16* hH  = (_Float16*)d_ws;            // N*64  f16
  _Float16* h4H = hH  + (size_t)NPTS * 64;    // N*128 f16
  _Float16* w5h = h4H + (size_t)NPTS * 128;   // 1024*192 f16
  float*   gmax = (float*)(w5h + 1024 * 192); // 1024 f32 (>=0, uint-max)

  kc_zero<<<4, 256, 0, stream>>>(gmax, 1024);
  kc_f32_to_f16<<<(1024 * 192) / 256, 256, 0, stream>>>(w5, w5h, 1024 * 192);

  const size_t ldsA = (size_t)8 * 2 * 16 * AS * sizeof(float);          // ~68 KB
  kc_point_mlp<<<NPTS / 128, 256, ldsA, stream>>>(x, w1, w2, w3, w4, hH, h4H);

  const size_t ldsC = (size_t)(8 * 16 * HCS2 + 2 * 16 * HCS2) * sizeof(_Float16)
                    + 1024 * sizeof(float);                              // ~68 KB
  kc_graph_gemm_max<<<NPTS / 128, 256, ldsC, stream>>>(hH, h4H, indices, w5h, gmax);

  kc_head<<<1, 512, 0, stream>>>(gmax, wf1, wf2, wf3, out);
}
